// HyperSplitMoE_65180423685434
// MI455X (gfx1250) — matrix operand order repaired and checked by reading, not compiled, not run on hardware
//
#include <hip/hip_runtime.h>
#include <hip/hip_bf16.h>

// ---------------------------------------------------------------------------
// Problem constants (fixed by the reference)
// ---------------------------------------------------------------------------
#define BB    16
#define CC    256
#define HH    64
#define WW    64
#define NPIX  (BB * HH * WW)      // 65536 pixels
#define CS    128                 // static channels
#define CD    128                 // dynamic channels
#define CE    256                 // expert expanded channels
#define OD    128
#define RR    16
#define EE    4

typedef __bf16 bf16;
typedef __attribute__((ext_vector_type(16))) __bf16 v16bf;
typedef __attribute__((ext_vector_type(8)))  __bf16 v8bf;
typedef __attribute__((ext_vector_type(8)))  float  v8f;
typedef __attribute__((ext_vector_type(4)))  unsigned int u32x4;
typedef __attribute__((ext_vector_type(8)))  int i32x8;
typedef __attribute__((ext_vector_type(4)))  int i32x4;

#if __has_builtin(__builtin_amdgcn_tensor_load_to_lds) && \
    __has_builtin(__builtin_amdgcn_s_wait_tensorcnt)
#define HAVE_TDM 1
#else
#define HAVE_TDM 0
#endif

static __device__ __forceinline__ bf16 f2bf(float f) {
    unsigned u = __float_as_uint(f);
    unsigned r = u + 0x7fffu + ((u >> 16) & 1u);   // RNE
    unsigned short h = (unsigned short)(r >> 16);
    return __builtin_bit_cast(bf16, h);
}
static __device__ __forceinline__ float bf2f(bf16 b) {
    unsigned short h = __builtin_bit_cast(unsigned short, b);
    return __uint_as_float(((unsigned)h) << 16);
}
static __device__ __forceinline__ float silu(float v) {
    return v / (1.0f + __expf(-v));
}

// ---------------------------------------------------------------------------
// Pack a [N][K] fp32 weight (out-major, as all reference 1x1 weights are)
// into WMMA-fragment-major bf16:  frag(kt,nt) -> 32 lanes x 16 contiguous bf16
// B order chosen here (contiguous k, 32x16): lane = half*16 + n ; element i -> K = half*16 + i
// ---------------------------------------------------------------------------
__global__ void pack_w_kernel(const float* __restrict__ src, bf16* __restrict__ dst,
                              int K, int N) {
    int idx = blockIdx.x * 256 + threadIdx.x;
    if (idx >= K * N) return;
    int k = idx / N, n = idx - k * N;
    int kt = k >> 5, kin = k & 31;
    int half = kin >> 4, i = kin & 15;
    int nt = n >> 4, nin = n & 15;
    int lane = half * 16 + nin;
    size_t o = (((size_t)kt * (N >> 4) + nt) * 32 + lane) * 16 + i;
    dst[o] = f2bf(src[(size_t)n * K + k]);
}

// ---------------------------------------------------------------------------
// NCHW fp32 -> pixel-major bf16  (x_pm[p*256 + c])  via LDS transpose
// ---------------------------------------------------------------------------
__global__ void nchw_to_pm_kernel(const float* __restrict__ x, bf16* __restrict__ xpm) {
    __shared__ float lds[32][65];
    int bh = blockIdx.x;                 // b*64 + h
    int cc = blockIdx.y * 32;
    int b = bh >> 6, h = bh & 63;
    const float* src = x + (((size_t)b * CC + cc) * (HH * WW)) + h * WW;
    for (int idx = threadIdx.x; idx < 32 * 64; idx += 256) {
        int c = idx >> 6, w = idx & 63;
        lds[c][w] = src[(size_t)c * (HH * WW) + w];
    }
    __syncthreads();
    bf16* dst = xpm + ((size_t)(b * (HH * WW) + h * WW)) * CC + cc;
    for (int idx = threadIdx.x; idx < 32 * 64; idx += 256) {
        int w = idx >> 5, c = idx & 31;
        dst[(size_t)w * CC + c] = f2bf(lds[c][w]);
    }
}

// ---------------------------------------------------------------------------
// g[b][c] = mean_{h,w} x[b, 128+c, h, w]
// ---------------------------------------------------------------------------
__global__ void mean_hw_kernel(const float* __restrict__ x, float* __restrict__ g) {
    int bc = blockIdx.x;
    int b = bc >> 7, c = bc & 127;
    const float* src = x + ((size_t)(b * CC + CD + c)) * (HH * WW);
    float s = 0.0f;
    for (int j = threadIdx.x; j < HH * WW; j += 256) s += src[j];
    __shared__ float red[256];
    red[threadIdx.x] = s;
    __syncthreads();
    for (int st = 128; st > 0; st >>= 1) {
        if (threadIdx.x < st) red[threadIdx.x] += red[threadIdx.x + st];
        __syncthreads();
    }
    if (threadIdx.x == 0) g[bc] = red[0] * (1.0f / (HH * WW));
}

// ---------------------------------------------------------------------------
// Router MLP + softmax + top-2 gate.  One thread per batch element.
// ---------------------------------------------------------------------------
__global__ void routing_kernel(const float* __restrict__ g,
                               const float* __restrict__ rw1, const float* __restrict__ rb1,
                               const float* __restrict__ rw2, const float* __restrict__ rb2,
                               float* __restrict__ gate) {
    int b = threadIdx.x;
    if (b >= BB) return;
    const float* gb = g + b * CD;
    float r[RR];
    for (int j = 0; j < RR; ++j) {
        float a = rb1[j];
        const float* wj = rw1 + j * CD;
        for (int c = 0; c < CD; ++c) a += gb[c] * wj[c];
        r[j] = silu(a);
    }
    float lg[EE];
    float m = -1e30f;
    for (int e = 0; e < EE; ++e) {
        float a = rb2[e];
        for (int j = 0; j < RR; ++j) a += r[j] * rw2[e * RR + j];
        lg[e] = a;
        m = fmaxf(m, a);
    }
    float p[EE], sum = 0.0f;
    for (int e = 0; e < EE; ++e) { p[e] = __expf(lg[e] - m); sum += p[e]; }
    for (int e = 0; e < EE; ++e) p[e] /= sum;
    int i1 = 0;
    for (int e = 1; e < EE; ++e) if (p[e] > p[i1]) i1 = e;
    int i2 = -1;
    for (int e = 0; e < EE; ++e) if (e != i1 && (i2 < 0 || p[e] > p[i2])) i2 = e;
    for (int e = 0; e < EE; ++e)
        gate[b * EE + e] = (e == i1 || e == i2) ? p[e] : 0.0f;
}

// ---------------------------------------------------------------------------
// Depthwise 3x3 ('SAME') + BN + SiLU on pixel-major bf16, 2 channels/thread.
// ---------------------------------------------------------------------------
__global__ void dwconv_pm_kernel(const bf16* __restrict__ in, int ldin, int C,
                                 const float* __restrict__ w9,
                                 const float* __restrict__ sc, const float* __restrict__ bs,
                                 bf16* __restrict__ out, int ldout,
                                 const float* __restrict__ gate, int expert) {
    int idx = blockIdx.x * 256 + threadIdx.x;
    int halfC = C >> 1;
    int p = idx / halfC;
    int cp = idx - p * halfC;
    if (p >= NPIX) return;
    int b = p >> 12;
    if (gate && gate[b * EE + expert] == 0.0f) return;
    int rem = p & 4095;
    int h = rem >> 6, w = rem & 63;
    int c0 = cp * 2;
    const float* wv = w9 + c0 * 9;
    float a0 = 0.0f, a1 = 0.0f;
    for (int dh = -1; dh <= 1; ++dh) {
        int hh = h + dh;
        if (hh < 0 || hh > 63) continue;
        for (int dw = -1; dw <= 1; ++dw) {
            int ww = w + dw;
            if (ww < 0 || ww > 63) continue;
            int q = p + dh * 64 + dw;
            const bf16* ip = in + (size_t)q * ldin + c0;
            int j = (dh + 1) * 3 + (dw + 1);
            a0 += bf2f(ip[0]) * wv[j];
            a1 += bf2f(ip[1]) * wv[9 + j];
        }
    }
    float y0 = silu(a0 * sc[c0] + bs[c0]);
    float y1 = silu(a1 * sc[c0 + 1] + bs[c0 + 1]);
    out[(size_t)p * ldout + c0]     = f2bf(y0);
    out[(size_t)p * ldout + c0 + 1] = f2bf(y1);
}

__global__ void zero_kernel(float* __restrict__ p) {
    p[blockIdx.x * 256 + threadIdx.x] = 0.0f;
}

// out_d_acc (fp32, pixel-major [p][128]) -> cat cols 128..255 (bf16)
__global__ void outd_to_cat_kernel(const float* __restrict__ acc, bf16* __restrict__ cat) {
    int idx = blockIdx.x * 256 + threadIdx.x;
    int p = idx >> 7, c = idx & 127;
    cat[(size_t)p * 256 + 128 + c] = f2bf(acc[idx]);
}

// ---------------------------------------------------------------------------
// bf16 WMMA GEMM: C[M x Ncols] = A[M x K] * B, block = 8 waves, wave = 32x32.
// A: pixel-major bf16 from global (2 contiguous 16B loads per frag per lane).
// B: frag-major packed bf16; the block's 64-column B panel (K/32 rows x 4KB,
//    row stride ntiles KB) is staged into LDS ONCE per block by the Tensor
//    Data Mover (tensor_load_to_lds, TENSORcnt), then all WMMA B fragments
//    are fed from LDS (ds_load_b128).
// MODE 0: out = silu(acc*scale[n]+bias[n]) -> bf16 outBf[p*ldo + n]
// MODE 2: out_d_acc[p*128+n] += gate * (acc*scale+bias + x_d[p,n])
// MODE 3: d_out NCHW = acc*scale+bias + x  (LDS-staged coalesced store)
// ---------------------------------------------------------------------------
#define SMEM_BYTES 33024   // max(B panel 32KB, 64x129 f32 output tile)

template <int MODE>
__global__ __launch_bounds__(256)
void gemm_bf16_wmma(const bf16* __restrict__ A, int lda,
                    const bf16* __restrict__ Bp, int K, int Ncols,
                    const float* __restrict__ scale, const float* __restrict__ bias,
                    bf16* __restrict__ outBf, int ldo,
                    float* __restrict__ outF,
                    const float* __restrict__ xres,
                    const float* __restrict__ gate, int expert) {
    __shared__ __align__(64) char smem[SMEM_BYTES];
    const int tid  = threadIdx.x;
    const int wave = tid >> 5;
    const int lane = tid & 31;
    const int wm   = wave & 3;        // 4 waves stacked along M
    const int wn   = wave >> 2;       // 2 waves along N
    const int pBase = blockIdx.x * 128;
    const int nBlk  = blockIdx.y * 64;
    const int bIdx  = pBase >> 12;    // all 128 rows share one batch element
    float gw = 1.0f;
    if (gate) {
        gw = gate[bIdx * EE + expert];
        if (gw == 0.0f) return;       // uniform across block: EXEC stays all-ones
    }
    const int ntiles = Ncols >> 4;
    const int nt0    = nBlk >> 4;     // first of this block's 4 n-tiles
    const int kRows  = K >> 5;        // B panel rows (one 4KB row per kt)

    // ---- Stage the B panel into LDS --------------------------------------
#if HAVE_TDM
    if (wave == 0) {
        unsigned long long gaddr =
            (unsigned long long)(uintptr_t)Bp + (unsigned long long)nt0 * 1024ull;
        unsigned ldsAddr = (unsigned)(uintptr_t)(void*)smem;
        unsigned td0 = 1024u;                         // 4KB row in 4B units
        unsigned td1 = (unsigned)kRows;
        unsigned long long st0 = (unsigned long long)ntiles * 256ull; // 4B units
        u32x4 g0;
        g0[0] = 1u;                                   // count=1, load descriptor
        g0[1] = ldsAddr;                              // lds_addr
        g0[2] = (unsigned)(gaddr & 0xffffffffull);    // global_addr[31:0]
        g0[3] = (unsigned)((gaddr >> 32) & 0x1ffffffull) | (2u << 30); // type=2
        i32x8 g1;
        g1[0] = (int)(2u << 16);                      // data_size = 4 bytes
        g1[1] = (int)((td0 & 0xffffu) << 16);         // tensor_dim0[15:0]
        g1[2] = (int)(((td0 >> 16) & 0xffffu) | ((td1 & 0xffffu) << 16));
        g1[3] = (int)(((td1 >> 16) & 0xffffu) | (td0 << 16));          // tile_dim0
        g1[4] = (int)(td1 & 0xffffu);                 // tile_dim1 (tile_dim2 = 0)
        g1[5] = (int)(st0 & 0xffffffffull);           // tensor_dim0_stride[31:0]
        g1[6] = (int)((st0 >> 32) & 0xffffull);       // stride[47:32], dim1_stride=0
        g1[7] = 0;
        i32x4 z4 = {0, 0, 0, 0};
#if defined(__clang_major__) && (__clang_major__ >= 23)
        i32x8 z8 = {0, 0, 0, 0, 0, 0, 0, 0};
        __builtin_amdgcn_tensor_load_to_lds(g0, g1, z4, z4, z8, 0);
#else
        __builtin_amdgcn_tensor_load_to_lds(g0, g1, z4, z4, 0);
#endif
        __builtin_amdgcn_s_wait_tensorcnt(0);
    }
#else
    {   // Fallback: cooperative vector copy global -> LDS
        unsigned* dstU = (unsigned*)smem;
        const unsigned* srcU = (const unsigned*)(Bp + (size_t)nt0 * 512);
        int totalU = kRows * 1024;                    // 4B units
        for (int u = tid; u < totalU; u += 256) {
            int kt = u >> 10, off = u & 1023;
            dstU[u] = srcU[(size_t)kt * ((size_t)ntiles << 8) + off];
        }
    }
#endif
    __syncthreads();

    const int half = lane >> 4;
    const int l16  = lane & 15;
    const int row0 = pBase + wm * 32 + l16;     // A rows for sub-tile 0
    const int row1 = row0 + 16;                 // A rows for sub-tile 1
    const int nW   = nBlk + wn * 32;

    v8f accs[2][2] = {{v8f{}, v8f{}}, {v8f{}, v8f{}}};
    const bf16* Arow0 = A + (size_t)row0 * lda;
    const bf16* Arow1 = A + (size_t)row1 * lda;
    const bf16* Bs = (const bf16*)smem;

    for (int kb = 0; kb < K; kb += 32) {
        const int kt = kb >> 5;
        v8bf a0lo = *(const v8bf*)(Arow0 + kb + half * 16);
        v8bf a0hi = *(const v8bf*)(Arow0 + kb + half * 16 + 8);
        v8bf a1lo = *(const v8bf*)(Arow1 + kb + half * 16);
        v8bf a1hi = *(const v8bf*)(Arow1 + kb + half * 16 + 8);
        v16bf a0 = __builtin_shufflevector(a0lo, a0hi, 0, 1, 2, 3, 4, 5, 6, 7,
                                           8, 9, 10, 11, 12, 13, 14, 15);
        v16bf a1 = __builtin_shufflevector(a1lo, a1hi, 0, 1, 2, 3, 4, 5, 6, 7,
                                           8, 9, 10, 11, 12, 13, 14, 15);
        // B fragments from LDS: frag (kt, wn*2 + bsub), 32B per lane
        v16bf b0 = *(const v16bf*)(Bs + ((size_t)(kt * 4 + wn * 2) << 9) + lane * 16);
        v16bf b1 = *(const v16bf*)(Bs + ((size_t)(kt * 4 + wn * 2 + 1) << 9) + lane * 16);
        accs[0][0] = __builtin_amdgcn_wmma_f32_16x16x32_bf16(false, a0, false, b0,
                                                             (short)0, accs[0][0], false, false);
        accs[0][1] = __builtin_amdgcn_wmma_f32_16x16x32_bf16(false, a0, false, b1,
                                                             (short)0, accs[0][1], false, false);
        accs[1][0] = __builtin_amdgcn_wmma_f32_16x16x32_bf16(false, a1, false, b0,
                                                             (short)0, accs[1][0], false, false);
        accs[1][1] = __builtin_amdgcn_wmma_f32_16x16x32_bf16(false, a1, false, b1,
                                                             (short)0, accs[1][1], false, false);
    }

    if constexpr (MODE == 3) {
        __syncthreads();   // B panel no longer needed; reuse smem as out tile
        float* ot = (float*)smem;   // [64][129]
#pragma unroll
        for (int asub = 0; asub < 2; ++asub)
#pragma unroll
            for (int bsub = 0; bsub < 2; ++bsub) {
                int n = nW + bsub * 16 + l16;
                float s = scale[n], bi = bias[n];
#pragma unroll
                for (int i = 0; i < 8; ++i) {
                    int pl = wm * 32 + asub * 16 + i + 8 * half;
                    int nl = n - nBlk;
                    ot[nl * 129 + pl] = accs[asub][bsub][i] * s + bi;
                }
            }
        __syncthreads();
        for (int it = tid; it < 64 * 128; it += 256) {
            int nl = it >> 7, pl = it & 127;
            int p = pBase + pl;
            int b = p >> 12, sp = p & 4095;
            size_t o = (((size_t)(b * CC + nBlk + nl)) << 12) + sp;
            outF[o] = ot[nl * 129 + pl] + xres[o];
        }
    } else {
#pragma unroll
        for (int asub = 0; asub < 2; ++asub)
#pragma unroll
            for (int bsub = 0; bsub < 2; ++bsub) {
                int n = nW + bsub * 16 + l16;
                float s = scale[n], bi = bias[n];
#pragma unroll
                for (int i = 0; i < 8; ++i) {
                    int p = pBase + wm * 32 + asub * 16 + i + 8 * half;
                    float v = accs[asub][bsub][i] * s + bi;
                    if constexpr (MODE == 0) {
                        outBf[(size_t)p * ldo + n] = f2bf(silu(v));
                    } else {  // MODE == 2: gated residual accumulate
                        int b = p >> 12, sp = p & 4095;
                        float xr = xres[(((size_t)(b * CC + CD + n)) << 12) + sp];
                        outF[(size_t)p * OD + n] += gw * (v + xr);
                    }
                }
            }
    }
}

// ---------------------------------------------------------------------------
// Host launcher
// ---------------------------------------------------------------------------
extern "C" void kernel_launch(void* const* d_in, const int* in_sizes, int n_in,
                              void* d_out, int out_size, void* d_ws, size_t ws_size,
                              hipStream_t stream) {
    const float* x      = (const float*)d_in[0];
    const float* sw_dw  = (const float*)d_in[1];
    const float* sbn1_s = (const float*)d_in[2];
    const float* sbn1_b = (const float*)d_in[3];
    const float* sw_pw  = (const float*)d_in[4];
    const float* sbn2_s = (const float*)d_in[5];
    const float* sbn2_b = (const float*)d_in[6];
    const float* rw1    = (const float*)d_in[7];
    const float* rb1    = (const float*)d_in[8];
    const float* rw2    = (const float*)d_in[9];
    const float* rb2    = (const float*)d_in[10];
    const float* ew_exp = (const float*)d_in[11];
    const float* e1s    = (const float*)d_in[12];
    const float* e1b    = (const float*)d_in[13];
    const float* ew_dw  = (const float*)d_in[14];
    const float* e2s    = (const float*)d_in[15];
    const float* e2b    = (const float*)d_in[16];
    const float* ew_proj= (const float*)d_in[17];
    const float* e3s    = (const float*)d_in[18];
    const float* e3b    = (const float*)d_in[19];
    const float* pw     = (const float*)d_in[20];
    const float* bn_s   = (const float*)d_in[21];
    const float* bn_b   = (const float*)d_in[22];
    float* out = (float*)d_out;
    (void)in_sizes; (void)n_in; (void)out_size; (void)ws_size;

    // Workspace carve-out (256B aligned)
    char* wsp = (char*)d_ws;
    auto carve = [&](size_t bytes) -> char* {
        char* r = wsp;
        wsp += (bytes + 255) & ~(size_t)255;
        return r;
    };
    float* g     = (float*)carve((size_t)BB * CD * 4);
    float* gate  = (float*)carve((size_t)BB * EE * 4);
    bf16*  Bpw   = (bf16*) carve((size_t)CS * 128 * 2);
    bf16*  Bfin  = (bf16*) carve((size_t)CC * CC * 2);
    bf16*  Bexp  = (bf16*) carve((size_t)EE * CD * CE * 2);
    bf16*  Bproj = (bf16*) carve((size_t)EE * CE * OD * 2);
    bf16*  xpm   = (bf16*) carve((size_t)NPIX * CC * 2);
    bf16*  hbf   = (bf16*) carve((size_t)NPIX * CS * 2);
    bf16*  tbuf  = (bf16*) carve((size_t)NPIX * CE * 2);
    bf16*  ubuf  = (bf16*) carve((size_t)NPIX * CE * 2);
    bf16*  cat   = (bf16*) carve((size_t)NPIX * CC * 2);
    float* outd  = (float*)carve((size_t)NPIX * OD * 4);

    // 1) Pack all 1x1 weights to WMMA-frag-major bf16
    pack_w_kernel<<<(CS * 128 + 255) / 256, 256, 0, stream>>>(sw_pw, Bpw, CS, 128);
    pack_w_kernel<<<(CC * CC + 255) / 256, 256, 0, stream>>>(pw, Bfin, CC, CC);
    for (int e = 0; e < EE; ++e) {
        pack_w_kernel<<<(CD * CE + 255) / 256, 256, 0, stream>>>(
            ew_exp + (size_t)e * CE * CD, Bexp + (size_t)e * CD * CE, CD, CE);
        pack_w_kernel<<<(CE * OD + 255) / 256, 256, 0, stream>>>(
            ew_proj + (size_t)e * OD * CE, Bproj + (size_t)e * CE * OD, CE, OD);
    }

    // 2) NCHW -> pixel-major bf16 (all 256 channels)
    nchw_to_pm_kernel<<<dim3(BB * HH, CC / 32), 256, 0, stream>>>(x, xpm);

    // 3) Routing
    mean_hw_kernel<<<BB * CD, 256, 0, stream>>>(x, g);
    routing_kernel<<<1, 32, 0, stream>>>(g, rw1, rb1, rw2, rb2, gate);

    // 4) Static branch: dw3x3+BN+SiLU, then 1x1 WMMA GEMM -> cat[:, 0:128]
    dwconv_pm_kernel<<<NPIX * (CS / 2) / 256, 256, 0, stream>>>(
        xpm, CC, CS, sw_dw, sbn1_s, sbn1_b, hbf, CS, nullptr, 0);
    gemm_bf16_wmma<0><<<dim3(NPIX / 128, 128 / 64), 256, 0, stream>>>(
        hbf, CS, Bpw, CS, 128, sbn2_s, sbn2_b, cat, CC, nullptr, nullptr, nullptr, 0);

    // 5) Experts (gate-skipped; at most top-2 per batch actually compute)
    zero_kernel<<<NPIX * OD / 256, 256, 0, stream>>>(outd);
    for (int e = 0; e < EE; ++e) {
        gemm_bf16_wmma<0><<<dim3(NPIX / 128, CE / 64), 256, 0, stream>>>(
            xpm + CD, CC, Bexp + (size_t)e * CD * CE, CD, CE,
            e1s + e * CE, e1b + e * CE, tbuf, CE, nullptr, nullptr, gate, e);
        dwconv_pm_kernel<<<NPIX * (CE / 2) / 256, 256, 0, stream>>>(
            tbuf, CE, CE, ew_dw + (size_t)e * CE * 9, e2s + e * CE, e2b + e * CE,
            ubuf, CE, gate, e);
        gemm_bf16_wmma<2><<<dim3(NPIX / 128, OD / 64), 256, 0, stream>>>(
            ubuf, CE, Bproj + (size_t)e * CE * OD, CE, OD,
            e3s + e * OD, e3b + e * OD, nullptr, 0, outd, x, gate, e);
    }
    outd_to_cat_kernel<<<NPIX * OD / 256, 256, 0, stream>>>(outd, cat);

    // 6) Final 1x1 (256->256) + BN + residual, coalesced NCHW store
    gemm_bf16_wmma<3><<<dim3(NPIX / 128, CC / 64), 256, 0, stream>>>(
        cat, CC, Bfin, CC, CC, bn_s, bn_b, nullptr, 0, out, x, nullptr, 0);
}
